// MHCLayerDynamic_20555713478902
// MI455X (gfx1250) — compile-verified, matrix operand order checked by reading, not run on hardware
//
#include <hip/hip_runtime.h>
#include <stdint.h>

#define B_TOT 8192
#define C_DIM 2048
#define NC    8192           // N*C = 4*2048
#define KTILE 256
#define KPAD  (KTILE + 8)    // pad to break LDS bank alignment (row stride 528B)
#define NWAVE 8
#define EPSF  1e-6f

typedef __attribute__((ext_vector_type(16))) __bf16   v16bf;
typedef __attribute__((ext_vector_type(8)))  float    v8f;
typedef __attribute__((ext_vector_type(4)))  uint32_t v4u;

union Frag { v16bf v; v4u q[2]; };

// Native RNE float->bf16 (backend picks hw cvt if available on gfx1250)
static __device__ __forceinline__ uint16_t f2bf(float f) {
  __bf16 h = (__bf16)f;
  return __builtin_bit_cast(uint16_t, h);
}
static __device__ __forceinline__ float bf2f(uint16_t h) {
  return __uint_as_float(((uint32_t)h) << 16);
}
static __device__ __forceinline__ float bfrt(float f) { return bf2f(f2bf(f)); }
static __device__ __forceinline__ float sigm(float z) { return 1.0f / (1.0f + __expf(-z)); }

__global__ __launch_bounds__(256)
void mhc_fused(const float* __restrict__ x, const float* __restrict__ w,
               const float* __restrict__ phi, const float* __restrict__ b_pre,
               const float* __restrict__ b_post, const float* __restrict__ b_res,
               float* __restrict__ out)
{
  __shared__ uint16_t x_s[16][KPAD];        // bf16 bits of x tile   (16 x 256)
  __shared__ uint16_t phi_s[32][KPAD];      // bf16 bits of phi tile (24 used, 8 zero)
  __shared__ float    red_s[NWAVE * 2 * 32 * 8]; // per-wave partial accumulators
  __shared__ float    scal_s[256];          // scalar reductions
  __shared__ float    H_s[16][24];          // raw H_proj tile
  __shared__ float    prm_s[16][24];        // per-row: pre[4], post[4], P[16]

  const int tid  = threadIdx.x;
  const int wave = tid >> 5;
  const int lane = tid & 31;
  const int half = lane >> 4;
  const int l16  = lane & 15;
  const size_t b0 = (size_t)blockIdx.x * 16;

  // zero phi_s pad rows 24..31 (never rewritten)
  for (int i = tid; i < 8 * KTILE; i += 256) phi_s[24 + (i >> 8)][i & 255] = 0;

  v8f c0 = {0.f, 0.f, 0.f, 0.f, 0.f, 0.f, 0.f, 0.f};
  v8f c1 = c0;
  float ssq = 0.f;

  const int xr  = tid >> 4;          // staging row (0..15)
  const int xc0 = (tid & 15) * 16;   // staging col base
  const int kb  = wave * 32;         // this wave's K slice inside the tile

  for (int kt = 0; kt < NC / KTILE; ++kt) {
    const int kbase = kt * KTILE;
    __syncthreads();
    // ---- stage x tile, f32 -> bf16, accumulate sumsq of rounded values ----
    {
      const float4* src = (const float4*)(x + (b0 + xr) * NC + kbase + xc0);
#pragma unroll
      for (int i = 0; i < 4; ++i) {
        float4 f = src[i];
        uint16_t h0 = f2bf(f.x), h1 = f2bf(f.y), h2 = f2bf(f.z), h3 = f2bf(f.w);
        float r0 = bf2f(h0), r1 = bf2f(h1), r2 = bf2f(h2), r3 = bf2f(h3);
        ssq += r0 * r0 + r1 * r1 + r2 * r2 + r3 * r3;
        int c = xc0 + 4 * i;
        *(uint32_t*)&x_s[xr][c]     = (uint32_t)h0 | ((uint32_t)h1 << 16);
        *(uint32_t*)&x_s[xr][c + 2] = (uint32_t)h2 | ((uint32_t)h3 << 16);
      }
    }
    // ---- stage phi tile rows 0..23 ----
#pragma unroll
    for (int i = 0; i < 6; ++i) {
      int ci = tid + 256 * i;            // 1536 chunks of 4 floats
      int r  = ci >> 6;                  // 0..23
      int c  = (ci & 63) * 4;
      float4 f = *(const float4*)(phi + (size_t)r * NC + kbase + c);
      *(uint32_t*)&phi_s[r][c]     = (uint32_t)f2bf(f.x) | ((uint32_t)f2bf(f.y) << 16);
      *(uint32_t*)&phi_s[r][c + 2] = (uint32_t)f2bf(f.z) | ((uint32_t)f2bf(f.w) << 16);
    }
    __syncthreads();
    // ---- WMMA on this wave's 32-wide K slice ----
    // 16-bit A 16x32 layout: lane(l16=M), elems 0..7 -> K = 8*half+0..7,
    //                        elems 8..15 -> K = 16+8*half+0..7  (two b128 LDS loads)
    Frag a, bf0, bf1;
    a.q[0]   = *(const v4u*)&x_s[l16][kb + 8 * half];
    a.q[1]   = *(const v4u*)&x_s[l16][kb + 16 + 8 * half];
    bf0.q[0] = *(const v4u*)&phi_s[l16][kb + 8 * half];
    bf0.q[1] = *(const v4u*)&phi_s[l16][kb + 16 + 8 * half];
    bf1.q[0] = *(const v4u*)&phi_s[16 + l16][kb + 8 * half];
    bf1.q[1] = *(const v4u*)&phi_s[16 + l16][kb + 16 + 8 * half];
    c0 = __builtin_amdgcn_wmma_f32_16x16x32_bf16(false, a.v, false, bf0.v, (short)0, c0, false, false);
    c1 = __builtin_amdgcn_wmma_f32_16x16x32_bf16(false, a.v, false, bf1.v, (short)0, c1, false, false);
  }

  // ---- spill partial accumulators + per-thread sumsq ----
#pragma unroll
  for (int e = 0; e < 8; ++e) {
    red_s[(((wave * 2 + 0) * 32 + lane) * 8) + e] = c0[e];
    red_s[(((wave * 2 + 1) * 32 + lane) * 8) + e] = c1[e];
  }
  scal_s[tid] = ssq;
  __syncthreads();

  float rinv = 0.f;
  if (tid < 16) {                          // per-row 1/rms_h
    float s = 0.f;
    for (int i = 0; i < 16; ++i) s += scal_s[tid * 16 + i];
    rinv = 1.0f / sqrtf(s / (float)NC + EPSF);
  }

  // ---- cross-wave reduction of C tiles into H_s ----
  for (int p = tid; p < 512; p += 256) {
    int tile = p >> 8, rem = p & 255, ln = rem >> 3, e = rem & 7;
    float s = 0.f;
#pragma unroll
    for (int wv = 0; wv < NWAVE; ++wv) s += red_s[(((wv * 2 + tile) * 32 + ln) * 8) + e];
    int o  = tile * 16 + (ln & 15);        // output column (C/D layout: N = lane%16)
    int mr = e + 8 * (ln >> 4);            // output row    (M = e + 8*(lane/16))
    if (o < 24) H_s[mr][o] = s;
  }
  __syncthreads();

  // ---- per-row gates + Sinkhorn (16 lanes of work) ----
  if (tid < 16) {
    const int m = tid;
    float hp[24];
#pragma unroll
    for (int i = 0; i < 24; ++i) hp[i] = H_s[m][i] * rinv;
#pragma unroll
    for (int i = 0; i < 4; ++i) {
      prm_s[m][i]     = sigm(0.1f * hp[i] + b_pre[i]);
      prm_s[m][4 + i] = 2.0f * sigm(0.1f * hp[4 + i] + b_post[i]);
    }
    float P[4][4];
#pragma unroll
    for (int i = 0; i < 4; ++i)
#pragma unroll
      for (int j = 0; j < 4; ++j)
        P[i][j] = __expf(0.1f * hp[8 + i * 4 + j] + b_res[i * 4 + j]);
#pragma unroll
    for (int it = 0; it < 3; ++it) {
      for (int i = 0; i < 4; ++i) {
        float r = 1.0f / (P[i][0] + P[i][1] + P[i][2] + P[i][3] + EPSF);
        P[i][0] *= r; P[i][1] *= r; P[i][2] *= r; P[i][3] *= r;
      }
      for (int j = 0; j < 4; ++j) {
        float r = 1.0f / (P[0][j] + P[1][j] + P[2][j] + P[3][j] + EPSF);
        P[0][j] *= r; P[1][j] *= r; P[2][j] *= r; P[3][j] *= r;
      }
    }
#pragma unroll
    for (int i = 0; i < 16; ++i) prm_s[m][8 + i] = P[i >> 2][i & 3];
  }
  __syncthreads();

  // ---- phase 2: aggregation + RMSNorm + mixing, per row ----
  const int cc = tid * 8;                  // 8 channels per thread
  float wb[8];
#pragma unroll
  for (int i = 0; i < 8; i += 4) {
    float4 f = *(const float4*)(w + cc + i);
    wb[i] = bfrt(f.x); wb[i + 1] = bfrt(f.y); wb[i + 2] = bfrt(f.z); wb[i + 3] = bfrt(f.w);
  }

  for (int m = 0; m < 16; ++m) {
    const float* xrow = x + (b0 + m) * NC;
    float xv[4][8];
#pragma unroll
    for (int n = 0; n < 4; ++n) {
      float4 f0 = *(const float4*)(xrow + n * C_DIM + cc);
      float4 f1 = *(const float4*)(xrow + n * C_DIM + cc + 4);
      xv[n][0] = f0.x; xv[n][1] = f0.y; xv[n][2] = f0.z; xv[n][3] = f0.w;
      xv[n][4] = f1.x; xv[n][5] = f1.y; xv[n][6] = f1.z; xv[n][7] = f1.w;
    }
    float p0 = prm_s[m][0], p1 = prm_s[m][1], p2 = prm_s[m][2], p3 = prm_s[m][3];
    float aggb[8];
    float ss = 0.f;
#pragma unroll
    for (int c = 0; c < 8; ++c) {
      float a  = p0 * xv[0][c] + p1 * xv[1][c] + p2 * xv[2][c] + p3 * xv[3][c];
      float ab = bfrt(a);                  // bf16 round-trip as in reference
      aggb[c] = ab;
      ss += ab * ab;
    }
    // wave32 shuffle reduction, then one cross-wave stage (2 barriers/row)
#pragma unroll
    for (int off = 16; off > 0; off >>= 1) ss += __shfl_down(ss, off, 32);
    if (lane == 0) scal_s[wave] = ss;
    __syncthreads();
    float tot = 0.f;
#pragma unroll
    for (int wv = 0; wv < NWAVE; ++wv) tot += scal_s[wv];
    float rmsi = 1.0f / sqrtf(tot / (float)C_DIM + EPSF);
    __syncthreads();

    float yn[8];
#pragma unroll
    for (int c = 0; c < 8; ++c) yn[c] = aggb[c] * rmsi * wb[c];

    float* orow = out + (b0 + m) * NC;
#pragma unroll
    for (int i = 0; i < 4; ++i) {
      float q0 = prm_s[m][8 + i * 4], q1 = prm_s[m][8 + i * 4 + 1];
      float q2 = prm_s[m][8 + i * 4 + 2], q3 = prm_s[m][8 + i * 4 + 3];
      float po = prm_s[m][4 + i];
      float ov[8];
#pragma unroll
      for (int c = 0; c < 8; ++c)
        ov[c] = q0 * xv[0][c] + q1 * xv[1][c] + q2 * xv[2][c] + q3 * xv[3][c] + po * yn[c];
      float4 o0 = {ov[0], ov[1], ov[2], ov[3]};
      float4 o1 = {ov[4], ov[5], ov[6], ov[7]};
      *(float4*)(orow + i * C_DIM + cc)     = o0;
      *(float4*)(orow + i * C_DIM + cc + 4) = o1;
    }
  }
}

extern "C" void kernel_launch(void* const* d_in, const int* in_sizes, int n_in,
                              void* d_out, int out_size, void* d_ws, size_t ws_size,
                              hipStream_t stream) {
  (void)in_sizes; (void)n_in; (void)out_size; (void)d_ws; (void)ws_size;
  const float* x      = (const float*)d_in[0];
  const float* w      = (const float*)d_in[1];
  const float* phi    = (const float*)d_in[2];
  const float* b_pre  = (const float*)d_in[3];
  const float* b_post = (const float*)d_in[4];
  const float* b_res  = (const float*)d_in[5];
  float* out = (float*)d_out;

  dim3 grid(B_TOT / 16);
  dim3 block(256);
  hipLaunchKernelGGL(mhc_fused, grid, block, 0, stream,
                     x, w, phi, b_pre, b_post, b_res, out);
}
